// NeighborhoodAttentionS2_21337397526733
// MI455X (gfx1250) — matrix operand order repaired and checked by reading, not compiled, not run on hardware
//
#include <hip/hip_runtime.h>
#include <hip/hip_bf16.h>
#include <cmath>

// ---------------- problem constants ----------------
#define NLAT    91
#define NLON    180
#define CH      512
#define NHEAD   8
#define HDIM    64            // CH / NHEAD
#define BATCH   2
#define HW      (NLAT * NLON) // 16380
#define NPAD    16384         // HW padded (multiple of 256)
#define SCALE_Q 0.044194173824159216f  // 1/sqrt(512)
#define MAXNBR  512           // max neighbors per latitude row (poles have 360)

typedef _Float16 f16;
typedef f16   v8h  __attribute__((ext_vector_type(8)));
typedef f16   v16h __attribute__((ext_vector_type(16)));
typedef float v8f  __attribute__((ext_vector_type(8)));

static __device__ __forceinline__ v16h combine16(v8h lo, v8h hi) {
    union { v16h v; v8h h[2]; } u;
    u.h[0] = lo; u.h[1] = hi;
    return u.v;
}

// ---------------------------------------------------------------------------
// 1) elementwise f32 -> f16 weight conversion
// ---------------------------------------------------------------------------
__global__ void cvt_w_kernel(const float* __restrict__ w, f16* __restrict__ o, int n) {
    int i = blockIdx.x * blockDim.x + threadIdx.x;
    int stride = gridDim.x * blockDim.x;
    for (; i < n; i += stride) o[i] = (f16)w[i];
}

// ---------------------------------------------------------------------------
// 2) LDS-tiled transpose + convert:  X[b][c][xy] (f32) -> Xt[b][xy][c] (f16)
//    xy padded to NPAD rows (zeros).  scale folded in (SCALE for query).
// ---------------------------------------------------------------------------
__global__ __launch_bounds__(256)
void transpose_cvt_kernel(const float* __restrict__ X, f16* __restrict__ Xt, float scale) {
    __shared__ float tile[32][33];
    const int xy0 = blockIdx.x * 32;
    const int c0  = blockIdx.y * 32;
    const int b   = blockIdx.z;
    const int tx  = threadIdx.x;       // 0..31
    const int ty  = threadIdx.y;       // 0..7
    #pragma unroll
    for (int r = 0; r < 4; ++r) {
        int c  = c0 + ty + 8 * r;
        int xy = xy0 + tx;
        float v = (xy < HW) ? X[((size_t)b * CH + c) * HW + xy] * scale : 0.0f;
        tile[ty + 8 * r][tx] = v;
    }
    __syncthreads();
    #pragma unroll
    for (int r = 0; r < 4; ++r) {
        int xy = xy0 + ty + 8 * r;
        int c  = c0 + tx;
        Xt[((size_t)b * NPAD + xy) * CH + c] = (f16)tile[tx][ty + 8 * r];
    }
}

// ---------------------------------------------------------------------------
// 3) CSR row-pointer build: rp[t] = lower_bound(seg_ho, t), t in [0,91]
// ---------------------------------------------------------------------------
__global__ void rowptr_kernel(const int* __restrict__ seg, int nnz, int* __restrict__ rp) {
    int t = threadIdx.x;
    if (t <= NLAT) {
        int lo = 0, hi = nnz;
        while (lo < hi) {
            int mid = (lo + hi) >> 1;
            if (seg[mid] < t) lo = mid + 1; else hi = mid;
        }
        rp[t] = lo;
    }
}

// ---------------------------------------------------------------------------
// 4) WMMA GEMM:  out = A(M=CH x K=CH) * Bm[b](N x K, row-major f16) + bias
//    Wave tile 32(M) x 64(N) = 8 accumulators, 128 v_wmma per wave,
//    double-buffered fragments so next k-step loads overlap current WMMAs.
//    Block = 8 waves -> 64(M) x 256(N) block tile.
//    NMAJOR=true : C[b][n][m]  (n-major, contiguous v8f store per lane,
//                               channel-contiguous reads for attention)
//    NMAJOR=false: C[b][m][n]  (final output layout (B,512,91,180))
// ---------------------------------------------------------------------------
template <bool NMAJOR>
__global__ __launch_bounds__(256)
void gemm_f16_kernel(const f16* __restrict__ A, const f16* __restrict__ Bm,
                     const float* __restrict__ bias, float* __restrict__ C) {
    const int lane = threadIdx.x & 31;
    const int wave = threadIdx.x >> 5;
    const int l16  = lane & 15;
    const int lg   = lane >> 4;            // 0 or 1
    const int b    = blockIdx.z;
    const int m0   = blockIdx.y * 64 + (wave >> 2) * 32;
    const int n0   = blockIdx.x * 256 + (wave & 3) * 64;

    const f16* __restrict__ Bb = Bm + (size_t)b * NPAD * CH;

    v8f acc[2][4];
    #pragma unroll
    for (int t = 0; t < 2; ++t)
        #pragma unroll
        for (int s = 0; s < 4; ++s)
            #pragma unroll
            for (int i = 0; i < 8; ++i) acc[t][s][i] = 0.0f;

    // double-buffered fragment registers
    v8h aLo[2][2], aHi[2][2];   // [buf][m-frag]
    v8h bLo[2][4], bHi[2][4];   // [buf][n-frag]

    // Fragment layouts per CDNA5 ISA 7.12.2:
    //  A lane: M = lane%16; halves 0..7  -> K = k0 + 8*(lane/16) + h
    //                       halves 8..15 -> K = k0 + 16 + 8*(lane/16) + (h-8)
    //  B lane: N = lane%16; same K mapping as A
    auto loadFrags = [&](int k0, int buf) {
        #pragma unroll
        for (int t = 0; t < 2; ++t) {
            const f16* pA = A + (size_t)(m0 + t * 16 + l16) * CH + k0 + 8 * lg;
            aLo[buf][t] = *(const v8h*)(pA);
            aHi[buf][t] = *(const v8h*)(pA + 16);
        }
        #pragma unroll
        for (int s = 0; s < 4; ++s) {
            const f16* pB = Bb + (size_t)(n0 + s * 16 + l16) * CH + k0 + 8 * lg;
            bLo[buf][s] = *(const v8h*)(pB);
            bHi[buf][s] = *(const v8h*)(pB + 16);
        }
    };

    loadFrags(0, 0);
    #pragma unroll
    for (int kk = 0; kk < CH / 32; ++kk) {
        const int cur = kk & 1;
        if (kk + 1 < CH / 32) loadFrags(32 * (kk + 1), cur ^ 1);
        #pragma unroll
        for (int t = 0; t < 2; ++t) {
            v16h av = combine16(aLo[cur][t], aHi[cur][t]);
            #pragma unroll
            for (int s = 0; s < 4; ++s) {
                v16h bv = combine16(bLo[cur][s], bHi[cur][s]);
                acc[t][s] = __builtin_amdgcn_wmma_f32_16x16x32_f16(
                    false, av, false, bv, (short)0, acc[t][s], false, false);
            }
        }
    }

    // store (C/D layout: lane 0-15 -> M rows 0..7, lanes 16-31 -> rows 8..15; N = lane%16)
    #pragma unroll
    for (int t = 0; t < 2; ++t) {
        const int mbase = m0 + t * 16 + 8 * lg;
        #pragma unroll
        for (int s = 0; s < 4; ++s) {
            const int n = n0 + s * 16 + l16;
            if (NMAJOR) {
                v8f out = acc[t][s];
                #pragma unroll
                for (int r = 0; r < 8; ++r) out[r] += bias[mbase + r];
                *reinterpret_cast<v8f*>(&C[((size_t)b * NPAD + n) * CH + mbase]) = out;
            } else if (n < HW) {
                #pragma unroll
                for (int r = 0; r < 8; ++r) {
                    int m = mbase + r;
                    C[((size_t)b * CH + m) * HW + n] = acc[t][s][r] + bias[m];
                }
            }
        }
    }
}

// ---------------------------------------------------------------------------
// 5) zero the 4 padding rows of yt (rows HW..NPAD-1), so final GEMM reads 0
// ---------------------------------------------------------------------------
__global__ void zero_pad_kernel(f16* __restrict__ yt) {
    int i = blockIdx.x * blockDim.x + threadIdx.x;  // 0 .. BATCH*4*CH-1
    int b = i / (4 * CH);
    int r = (i / CH) & 3;
    int c = i % CH;
    yt[((size_t)b * NPAD + HW + r) * CH + c] = (f16)0.0f;
}

// ---------------------------------------------------------------------------
// 6) spherical neighborhood attention (n-major q/k/v: [b][n][c], n = x*NLON+y)
//    grid = (NLON, NLAT, BATCH), block = 256 = 8 waves, wave w == head w.
//    Pass 1: lanes parallelize neighbors -> logits (float4 dot over 64 ch,
//            q cached in LDS), wave shuffle-reduce max.
//    Pass 2: softmax-weighted V accumulation, 2 contiguous channels per lane
//            (coalesced b64 per neighbor row).
//    Output written transposed + f16: yt[b][n][h*64+c]
// ---------------------------------------------------------------------------
__global__ __launch_bounds__(256)
void attention_kernel(const float* __restrict__ qp, const float* __restrict__ kp,
                      const float* __restrict__ vp, const float* __restrict__ quad,
                      const int* __restrict__ rp, const int* __restrict__ hi_idx,
                      const int* __restrict__ wi_idx, f16* __restrict__ yt) {
    __shared__ float sLog[NHEAD][MAXNBR];
    __shared__ float sQ[NHEAD][HDIM];

    const int wo   = blockIdx.x;
    const int segi = blockIdx.y;
    const int b    = blockIdx.z;
    const int h    = threadIdx.x >> 5;
    const int lane = threadIdx.x & 31;

    const int jbeg  = rp[segi];
    const int count = rp[segi + 1] - jbeg;

    const size_t rowq = ((size_t)b * NPAD + (size_t)segi * NLON + wo) * CH + h * HDIM;

    // cache this (b,h,seg,wo) query vector in LDS (2 contiguous channels/lane)
    sQ[h][2 * lane]     = qp[rowq + 2 * lane];
    sQ[h][2 * lane + 1] = qp[rowq + 2 * lane + 1];
    __syncthreads();

    // ---- pass 1: logits + max --------------------------------------------
    float mymax = -INFINITY;
    for (int j0 = 0; j0 < count; j0 += 32) {
        int j = j0 + lane;
        float logit = -INFINITY;
        if (j < count) {
            int hi = hi_idx[jbeg + j];
            int wi = wi_idx[jbeg + j] + wo;
            if (wi >= NLON) wi -= NLON;
            const float4* k4 =
                reinterpret_cast<const float4*>(kp + ((size_t)b * NPAD + (size_t)hi * NLON + wi) * CH + h * HDIM);
            float acc = 0.0f;
            #pragma unroll
            for (int c4 = 0; c4 < HDIM / 4; ++c4) {
                float4 kv = k4[c4];
                float4 qv = *reinterpret_cast<const float4*>(&sQ[h][4 * c4]);
                acc += kv.x * qv.x + kv.y * qv.y + kv.z * qv.z + kv.w * qv.w;
            }
            sLog[h][j] = acc;
            logit = acc;
        }
        mymax = fmaxf(mymax, logit);
    }
    #pragma unroll
    for (int off = 16; off > 0; off >>= 1)
        mymax = fmaxf(mymax, __shfl_xor(mymax, off, 32));

    // ---- pass 2: softmax-weighted V accumulation -------------------------
    float num0 = 0.0f, num1 = 0.0f, den = 0.0f;
    for (int j = 0; j < count; ++j) {
        int hi = hi_idx[jbeg + j];
        int wi = wi_idx[jbeg + j] + wo;
        if (wi >= NLON) wi -= NLON;
        float e = __expf(sLog[h][j] - mymax) * quad[hi];
        const float* vrow = vp + ((size_t)b * NPAD + (size_t)hi * NLON + wi) * CH + h * HDIM + 2 * lane;
        num0 += e * vrow[0];
        num1 += e * vrow[1];
        den  += e;
    }
    float inv = 1.0f / den;
    size_t orow = ((size_t)b * NPAD + (size_t)segi * NLON + wo) * CH + h * HDIM + 2 * lane;
    yt[orow]     = (f16)(num0 * inv);
    yt[orow + 1] = (f16)(num1 * inv);
}

// ---------------------------------------------------------------------------
// launch
// ---------------------------------------------------------------------------
extern "C" void kernel_launch(void* const* d_in, const int* in_sizes, int n_in,
                              void* d_out, int out_size, void* d_ws, size_t ws_size,
                              hipStream_t stream) {
    const float* query  = (const float*)d_in[0];
    const float* key_in = (const float*)d_in[1];
    const float* value  = (const float*)d_in[2];
    const float* q_w    = (const float*)d_in[3];
    const float* k_w    = (const float*)d_in[4];
    const float* v_w    = (const float*)d_in[5];
    const float* proj_w = (const float*)d_in[6];
    const float* q_b    = (const float*)d_in[7];
    const float* k_b    = (const float*)d_in[8];
    const float* v_b    = (const float*)d_in[9];
    const float* proj_b = (const float*)d_in[10];
    const float* quad   = (const float*)d_in[11];
    const int*   seg    = (const int*)d_in[12];
    const int*   hi_i   = (const int*)d_in[13];
    const int*   wi_i   = (const int*)d_in[14];
    const int    nnz    = in_sizes[12];

    // ---- workspace carve-out ----
    char* ws = (char*)d_ws;
    size_t off = 0;
    auto carve = [&](size_t bytes) -> char* {
        char* p = ws + off;
        off += (bytes + 255) & ~(size_t)255;
        return p;
    };
    const size_t wbytes  = (size_t)CH * CH * sizeof(f16);           // 512 KB
    const size_t xtbytes = (size_t)BATCH * NPAD * CH * sizeof(f16); // 32 MB
    const size_t pbytes  = (size_t)BATCH * NPAD * CH * sizeof(float); // 64 MB (n-major)

    f16*   Wq16 = (f16*)carve(wbytes);
    f16*   Wk16 = (f16*)carve(wbytes);
    f16*   Wv16 = (f16*)carve(wbytes);
    f16*   Wp16 = (f16*)carve(wbytes);
    f16*   Xtq  = (f16*)carve(xtbytes);
    f16*   Xtk  = (f16*)carve(xtbytes);
    f16*   Xtv  = (f16*)carve(xtbytes);
    float* qp   = (float*)carve(pbytes);
    float* kp   = (float*)carve(pbytes);
    float* vp   = (float*)carve(pbytes);
    int*   rptr = (int*)carve((NLAT + 1) * sizeof(int));
    f16*   yt   = Xtq;   // reuse: query activations dead after q-projection GEMM

    // ---- 1) weights -> f16 ----
    cvt_w_kernel<<<dim3(256), 256, 0, stream>>>(q_w,    Wq16, CH * CH);
    cvt_w_kernel<<<dim3(256), 256, 0, stream>>>(k_w,    Wk16, CH * CH);
    cvt_w_kernel<<<dim3(256), 256, 0, stream>>>(v_w,    Wv16, CH * CH);
    cvt_w_kernel<<<dim3(256), 256, 0, stream>>>(proj_w, Wp16, CH * CH);

    // ---- 2) activations: transpose + convert (SCALE folded into query) ----
    dim3 tg(NPAD / 32, CH / 32, BATCH), tb(32, 8);
    transpose_cvt_kernel<<<tg, tb, 0, stream>>>(query,  Xtq, SCALE_Q);
    transpose_cvt_kernel<<<tg, tb, 0, stream>>>(key_in, Xtk, 1.0f);
    transpose_cvt_kernel<<<tg, tb, 0, stream>>>(value,  Xtv, 1.0f);

    // ---- 3) CSR row pointers ----
    rowptr_kernel<<<1, 128, 0, stream>>>(seg, nnz, rptr);

    // ---- 4) q/k/v projection GEMMs (WMMA), n-major output for attention ----
    dim3 gg(NPAD / 256, CH / 64, BATCH);
    gemm_f16_kernel<true><<<gg, 256, 0, stream>>>(Wq16, Xtq, q_b, qp);
    gemm_f16_kernel<true><<<gg, 256, 0, stream>>>(Wk16, Xtk, k_b, kp);
    gemm_f16_kernel<true><<<gg, 256, 0, stream>>>(Wv16, Xtv, v_b, vp);

    // ---- 5) zero yt pad rows (yt aliases Xtq; safe after q GEMM) ----
    zero_pad_kernel<<<dim3((BATCH * 4 * CH) / 256), 256, 0, stream>>>(yt);

    // ---- 6) neighborhood attention -> yt (f16, n-major) ----
    attention_kernel<<<dim3(NLON, NLAT, BATCH), 256, 0, stream>>>(
        qp, kp, vp, quad, rptr, hi_i, wi_i, yt);

    // ---- 7) output projection GEMM (WMMA) -> d_out, m-major (B,512,91,180) ----
    gemm_f16_kernel<false><<<gg, 256, 0, stream>>>(Wp16, yt, proj_b, (float*)d_out);

    (void)n_in; (void)out_size; (void)ws_size;
}
